// ExpertTransformer_15350213116676
// MI455X (gfx1250) — matrix operand order repaired and checked by reading, not compiled, not run on hardware
//
#include <hip/hip_runtime.h>
#include <hip/hip_bf16.h>
#include <cmath>

// ---------------------------------------------------------------------------
// Model dims (from reference)
// ---------------------------------------------------------------------------
#define BATCH 4
#define TSEQ  512
#define KWN   32
#define NPV   196
#define HID   512
#define NHD   8
#define DH    64
#define HH    32
#define ENCD  2048
#define FCD   2048
#define VOC   32000
#define NLAY  4
#define LNEPS 1e-5f

typedef __attribute__((ext_vector_type(16))) _Float16 v16h;
typedef __attribute__((ext_vector_type(8)))  float    v8f;

// ---------------------------------------------------------------------------
// Generic WMMA GEMM: C[M,N] = epi(A[M,K] * B)  with B(k,n)=B[k*ldbK + n*ldbN]
// Block tile 64x64x32, 256 threads = 8 waves; each wave computes a 16x32
// strip: one A fragment reused across two B fragments (2 WMMA accumulators).
// All K dims are multiples of 32 and all N multiples of 64 in this model, so
// float4 (b128) staging loads need only M-guards.
// EPI: 0 = none, 1 = +bias[n], 2 = exact GELU.   BT: 1 = B stored K-contiguous
// (e.g. We^T for the lm-head) -> stage with K-major coalescing.
// ---------------------------------------------------------------------------
#define BM 64
#define BN 64
#define BK 32
#define GTHREADS 256

template<int EPI, int BT>
__global__ __launch_bounds__(GTHREADS)
void gemm_wmma_kernel(const float* __restrict__ A, const float* __restrict__ B,
                      const float* __restrict__ bias, float* __restrict__ C,
                      int M, int N, int Kd, int lda, long ldbK, long ldbN, int ldc)
{
    __shared__ _Float16 As[BM][BK + 8];
    __shared__ _Float16 Bs[BK][BN + 8];

    const int tid  = threadIdx.x;
    const int lane = tid & 31;
    const int wave = tid >> 5;
    const int m0   = blockIdx.y * BM;
    const int n0   = blockIdx.x * BN;
    const int sm   = (wave & 3) * 16;   // wave's M offset within block tile
    const int sn   = (wave >> 2) * 32;  // wave's N offset (2 x 16 subtiles)
    const int mloc = lane & 15;
    const int half = lane >> 4;

    v8f acc0 = {}, acc1 = {};

    for (int k0 = 0; k0 < Kd; k0 += BK) {
        // ---- stage A tile (64x32) : float4 global loads, fp32 -> f16 ----
        #pragma unroll
        for (int i = 0; i < 2; ++i) {
            int idx = tid + i * GTHREADS;     // 0..511
            int r   = idx >> 3;               // 0..63
            int c4  = (idx & 7) * 4;          // 0,4,..,28
            int gm  = m0 + r;
            float4 v = make_float4(0.f, 0.f, 0.f, 0.f);
            if (gm < M) v = *(const float4*)&A[(long)gm * lda + (k0 + c4)];
            As[r][c4 + 0] = (_Float16)v.x;
            As[r][c4 + 1] = (_Float16)v.y;
            As[r][c4 + 2] = (_Float16)v.z;
            As[r][c4 + 3] = (_Float16)v.w;
        }
        // ---- stage B tile (32x64) ----
        if (BT == 0) {
            // row-major weights [K,N]: coalesced along N
            #pragma unroll
            for (int i = 0; i < 2; ++i) {
                int idx = tid + i * GTHREADS;
                int r   = idx >> 4;           // k: 0..31
                int c4  = (idx & 15) * 4;     // n: 0,4,..,60
                int gk  = k0 + r, gn = n0 + c4;
                float4 v = make_float4(0.f, 0.f, 0.f, 0.f);
                if (gn + 3 < N) v = *(const float4*)&B[(long)gk * ldbK + gn];
                Bs[r][c4 + 0] = (_Float16)v.x;
                Bs[r][c4 + 1] = (_Float16)v.y;
                Bs[r][c4 + 2] = (_Float16)v.z;
                Bs[r][c4 + 3] = (_Float16)v.w;
            }
        } else {
            // K-contiguous B (We^T): coalesced along K, scatter into LDS
            #pragma unroll
            for (int i = 0; i < 2; ++i) {
                int idx = tid + i * GTHREADS;
                int n   = idx >> 3;           // 0..63
                int k4  = (idx & 7) * 4;      // 0,4,..,28
                int gn  = n0 + n, gk = k0 + k4;
                float4 v = make_float4(0.f, 0.f, 0.f, 0.f);
                if (gn < N) v = *(const float4*)&B[(long)gn * ldbN + gk];
                Bs[k4 + 0][n] = (_Float16)v.x;
                Bs[k4 + 1][n] = (_Float16)v.y;
                Bs[k4 + 2][n] = (_Float16)v.z;
                Bs[k4 + 3][n] = (_Float16)v.w;
            }
        }
        __syncthreads();

        if (k0 + BK < Kd) {  // speculative prefetch of next A tile
            __builtin_prefetch(&A[(long)(m0 + (tid & 63)) * lda + (k0 + BK)], 0, 1);
        }

        // A fragment 16x32 f16 (lanes0-15: K=0..15, lanes16-31: K=16..31)
        v16h af;
        #pragma unroll
        for (int v = 0; v < 8; ++v) {
            int k = half * 16 + 2 * v;
            af[2 * v]     = As[sm + mloc][k];
            af[2 * v + 1] = As[sm + mloc][k + 1];
        }
        // B fragments 32x16 f16 (lanes0-15: K=e, lanes16-31: K=16+e; N=lane&15)
        v16h bf0, bf1;
        #pragma unroll
        for (int e = 0; e < 16; ++e) {
            bf0[e] = Bs[half * 16 + e][sn + mloc];
            bf1[e] = Bs[half * 16 + e][sn + 16 + mloc];
        }
        acc0 = __builtin_amdgcn_wmma_f32_16x16x32_f16(false, af, false, bf0,
                                                      (short)0, acc0, false, false);
        acc1 = __builtin_amdgcn_wmma_f32_16x16x32_f16(false, af, false, bf1,
                                                      (short)0, acc1, false, false);
        __syncthreads();
    }

    // epilogue + store (C layout: reg r -> row r + 8*half, col = lane&15)
    #pragma unroll
    for (int r = 0; r < 8; ++r) {
        int gm = m0 + sm + r + half * 8;
        if (gm < M) {
            #pragma unroll
            for (int j = 0; j < 2; ++j) {
                int gn = n0 + sn + j * 16 + mloc;
                if (gn < N) {
                    float v = (j == 0) ? acc0[r] : acc1[r];
                    if (EPI == 1) v += bias[gn];
                    if (EPI == 2) v = 0.5f * v * (1.f + erff(v * 0.70710678118f));
                    C[(long)gm * ldc + gn] = v;
                }
            }
        }
    }
}

// ---------------------------------------------------------------------------
// Fused differential attention.
// One wave per (batch, head-pair, 16-query tile). Does:
//   s1 = q1 k1^T/sqrt(HH), s2 = q2 k2^T/sqrt(HH)  (WMMA, masked)
//   attn = softmax(s1) - lam * softmax(s2)
//   out  = RMSnorm(attn @ v) * rms_w * (1 - lam_init)
// ---------------------------------------------------------------------------
#define TKMAX 512

__global__ __launch_bounds__(32)
void diff_attn_kernel(const float* __restrict__ Q, const float* __restrict__ Km,
                      const float* __restrict__ Vm,
                      const float* __restrict__ lq1, const float* __restrict__ lq2,
                      const float* __restrict__ lk1, const float* __restrict__ lk2,
                      const float* __restrict__ rmsw,
                      float* __restrict__ Out,
                      int Tq, int Tk, int causal, float lam_init)
{
    __shared__ float     sc1[16][TKMAX];
    __shared__ float     sc2[16][TKMAX];
    __shared__ _Float16  atw[16][TKMAX];

    const int lane = threadIdx.x;
    const int b    = blockIdx.z;
    const int h    = blockIdx.y;
    const int t0   = blockIdx.x * 16;
    const int mloc = lane & 15;
    const int half = lane >> 4;
    const float scale = 0.17677669529f;   // 1/sqrt(HH=32)

    // lambda = exp(lq1.lk1) - exp(lq2.lk2) + lam_init   (HH=32 -> one elem/lane)
    float p1 = lq1[lane] * lk1[lane];
    float p2 = lq2[lane] * lk2[lane];
    #pragma unroll
    for (int m = 16; m >= 1; m >>= 1) {
        p1 += __shfl_xor(p1, m, 32);
        p2 += __shfl_xor(p2, m, 32);
    }
    const float lam = expf(p1) - expf(p2) + lam_init;

    // Q fragments for head pair (16x32, K = HH exactly)
    v16h aq1, aq2;
    {
        int trow = t0 + mloc; if (trow >= Tq) trow = Tq - 1;
        const float* qp = Q + ((long)(b * Tq + trow)) * HID + (2 * h) * HH;
        #pragma unroll
        for (int v = 0; v < 8; ++v) {
            int k = half * 16 + 2 * v;
            aq1[2 * v]     = (_Float16)qp[k];
            aq1[2 * v + 1] = (_Float16)qp[k + 1];
            aq2[2 * v]     = (_Float16)qp[HH + k];
            aq2[2 * v + 1] = (_Float16)qp[HH + k + 1];
        }
    }

    // ---- scores: s = q k^T (WMMA over 16-key tiles) ----
    for (int n0 = 0; n0 < Tk; n0 += 16) {
        v16h bk1, bk2;
        {
            int nrow = n0 + mloc; if (nrow >= Tk) nrow = Tk - 1;
            const float* kp = Km + ((long)(b * Tk + nrow)) * HID + (2 * h) * HH;
            #pragma unroll
            for (int e = 0; e < 16; ++e) {
                int kk = half * 16 + e;
                bk1[e] = (_Float16)kp[kk];
                bk2[e] = (_Float16)kp[HH + kk];
            }
        }
        v8f z = {};
        v8f r1 = __builtin_amdgcn_wmma_f32_16x16x32_f16(false, aq1, false, bk1, (short)0, z, false, false);
        v8f r2 = __builtin_amdgcn_wmma_f32_16x16x32_f16(false, aq2, false, bk2, (short)0, z, false, false);
        #pragma unroll
        for (int r = 0; r < 8; ++r) {
            int row = r + half * 8;
            int tq  = t0 + row;
            int n   = n0 + mloc;
            float v1 = r1[r] * scale, v2 = r2[r] * scale;
            bool ok = (n < Tk) && (!causal || (n <= tq));
            if (!ok) { v1 = -3.0e38f; v2 = -3.0e38f; }
            sc1[row][n] = v1;
            sc2[row][n] = v2;
        }
    }
    __syncthreads();

    // ---- softmax both branches + differential combine into f16 attn ----
    for (int r = 0; r < 16; ++r) {
        float mx1 = -3.0e38f, mx2 = -3.0e38f;
        for (int c = lane; c < Tk; c += 32) {
            mx1 = fmaxf(mx1, sc1[r][c]);
            mx2 = fmaxf(mx2, sc2[r][c]);
        }
        #pragma unroll
        for (int m = 16; m >= 1; m >>= 1) {
            mx1 = fmaxf(mx1, __shfl_xor(mx1, m, 32));
            mx2 = fmaxf(mx2, __shfl_xor(mx2, m, 32));
        }
        float sm1 = 0.f, sm2 = 0.f;
        for (int c = lane; c < Tk; c += 32) {
            float e1 = expf(sc1[r][c] - mx1);
            float e2 = expf(sc2[r][c] - mx2);
            sc1[r][c] = e1; sc2[r][c] = e2;
            sm1 += e1; sm2 += e2;
        }
        #pragma unroll
        for (int m = 16; m >= 1; m >>= 1) {
            sm1 += __shfl_xor(sm1, m, 32);
            sm2 += __shfl_xor(sm2, m, 32);
        }
        float i1 = 1.f / fmaxf(sm1, 1e-30f);
        float i2 = 1.f / fmaxf(sm2, 1e-30f);
        for (int c = lane; c < Tk; c += 32)
            atw[r][c] = (_Float16)(sc1[r][c] * i1 - lam * sc2[r][c] * i2);
    }
    __syncthreads();

    // ---- out = attn @ v  (K-dim = Tk in 32-chunks; N = DH = 64 -> 4 tiles) ----
    v8f oacc[4] = { {}, {}, {}, {} };
    for (int kk0 = 0; kk0 < Tk; kk0 += 32) {
        v16h aa;
        #pragma unroll
        for (int v = 0; v < 8; ++v) {
            int k  = half * 16 + 2 * v;
            int c0 = kk0 + k;
            aa[2 * v]     = (c0     < Tk) ? atw[mloc][c0]     : (_Float16)0.f;
            aa[2 * v + 1] = (c0 + 1 < Tk) ? atw[mloc][c0 + 1] : (_Float16)0.f;
        }
        #pragma unroll
        for (int j = 0; j < 4; ++j) {
            v16h bv;
            #pragma unroll
            for (int e = 0; e < 16; ++e) {
                int krow = kk0 + half * 16 + e;
                float v = (krow < Tk)
                    ? Vm[((long)(b * Tk + krow)) * HID + h * DH + j * 16 + mloc] : 0.f;
                bv[e] = (_Float16)v;
            }
            oacc[j] = __builtin_amdgcn_wmma_f32_16x16x32_f16(false, aa, false, bv,
                                                             (short)0, oacc[j], false, false);
        }
    }

    // ---- per-row RMS norm over DH, * rms_w, * (1-lam_init), store ----
    const float oscale = 1.0f - lam_init;
    #pragma unroll
    for (int r = 0; r < 8; ++r) {
        float ss = 0.f;
        #pragma unroll
        for (int j = 0; j < 4; ++j) { float v = oacc[j][r]; ss += v * v; }
        ss += __shfl_xor(ss, 1, 32);
        ss += __shfl_xor(ss, 2, 32);
        ss += __shfl_xor(ss, 4, 32);
        ss += __shfl_xor(ss, 8, 32);
        float inv = rsqrtf(ss * (1.0f / DH) + LNEPS);
        int tq = t0 + r + half * 8;
        if (tq < Tq) {
            #pragma unroll
            for (int j = 0; j < 4; ++j) {
                float v = oacc[j][r] * inv * rmsw[j * 16 + mloc] * oscale;
                Out[((long)(b * Tq + tq)) * HID + h * DH + j * 16 + mloc] = v;
            }
        }
    }
}

// ---------------------------------------------------------------------------
// x = LayerNorm(x + y) * w + b      (one row per block, 128 threads)
// ---------------------------------------------------------------------------
__global__ __launch_bounds__(128)
void add_ln_kernel(float* __restrict__ x, const float* __restrict__ y,
                   const float* __restrict__ w, const float* __restrict__ bb)
{
    __shared__ float red[8];
    const int row = blockIdx.x, tid = threadIdx.x;
    const long base = (long)row * HID;
    float v[4];
    float s = 0.f;
    #pragma unroll
    for (int i = 0; i < 4; ++i) {
        int c = tid + i * 128;
        v[i] = x[base + c] + y[base + c];
        s += v[i];
    }
    #pragma unroll
    for (int m = 16; m >= 1; m >>= 1) s += __shfl_xor(s, m, 32);
    if ((tid & 31) == 0) red[tid >> 5] = s;
    __syncthreads();
    s = red[0] + red[1] + red[2] + red[3];
    const float mean = s * (1.0f / HID);
    float vs = 0.f;
    #pragma unroll
    for (int i = 0; i < 4; ++i) { float d = v[i] - mean; vs += d * d; }
    #pragma unroll
    for (int m = 16; m >= 1; m >>= 1) vs += __shfl_xor(vs, m, 32);
    if ((tid & 31) == 0) red[4 + (tid >> 5)] = vs;
    __syncthreads();
    vs = red[4] + red[5] + red[6] + red[7];
    const float inv = rsqrtf(vs * (1.0f / HID) + LNEPS);
    #pragma unroll
    for (int i = 0; i < 4; ++i) {
        int c = tid + i * 128;
        x[base + c] = (v[i] - mean) * inv * w[c] + bb[c];
    }
}

// ---------------------------------------------------------------------------
// Embedding gathers
// ---------------------------------------------------------------------------
__global__ void embed_kw_kernel(const int* __restrict__ tok, const float* __restrict__ We,
                                float* __restrict__ out, int rows)
{
    long i = (long)blockIdx.x * blockDim.x + threadIdx.x;
    if (i >= (long)rows * HID) return;
    int row = (int)(i / HID), col = (int)(i % HID);
    out[i] = We[(long)tok[row] * HID + col];
}

__global__ void embed_tok_kernel(const int* __restrict__ tok, const float* __restrict__ We,
                                 const float* __restrict__ wpe, float* __restrict__ out, int rows)
{
    long i = (long)blockIdx.x * blockDim.x + threadIdx.x;
    if (i >= (long)rows * HID) return;
    int row = (int)(i / HID), col = (int)(i % HID);
    int t = row % TSEQ;
    out[i] = We[(long)tok[row] * HID + col] + wpe[(long)t * HID + col];
}

// ---------------------------------------------------------------------------
// Host driver
// ---------------------------------------------------------------------------
extern "C" void kernel_launch(void* const* d_in, const int* in_sizes, int n_in,
                              void* d_out, int out_size, void* d_ws, size_t ws_size,
                              hipStream_t stream)
{
    (void)in_sizes; (void)n_in; (void)out_size; (void)ws_size;

    const float* vis    = (const float*)d_in[0];
    const int*   tokens = (const int*)d_in[1];
    const int*   kwtok  = (const int*)d_in[2];
    auto F = [&](int i) { return (const float*)d_in[i]; };

    // jax pytree flatten order (dict keys sorted):
    // params: We, dec[4], fuser[4], lang_enc, vf_b, vf_w, wpe
    // mha: lk1,lk2,lq1,lq2,rms_w,wk,wo,wq,wv ; ln: b,w ; mlp: fc,proj
    const int P0      = 3;
    const int IDX_We  = P0;
    const int DEC0    = P0 + 1;            // stride 26: ca@0 ln1@9 ln2@11 ln3@13 mlp@15 sa@17
    const int FUS0    = P0 + 1 + 104;      // stride 15: attn@0 ln1@9 ln2@11 mlp@13
    const int LANG    = P0 + 165;
    const int IDX_vfb = P0 + 174;
    const int IDX_vfw = P0 + 175;
    const int IDX_wpe = P0 + 176;

    // workspace carve-up (floats)
    float* ws = (float*)d_ws;
    float* kw = ws;                          // B*KW*HID    = 65536
    float* ef = kw + 65536;                  // B*NP*HID    = 401408
    float* x  = ef + 401408;                 // B*T*HID     = 1048576
    float* qb = x  + 1048576;                // 1048576
    float* kb = qb + 1048576;                // 1048576
    float* vb = kb + 1048576;                // 1048576
    float* ao = vb + 1048576;                // 1048576
    float* t1 = ao + 1048576;                // rows*FC max = 4194304
    float* t2 = t1 + 4194304;                // 1048576

    auto gemm = [&](const float* A, const float* Bm, const float* bias, float* C,
                    int M, int N, int Kd, long ldbK, long ldbN, int epi, int bt) {
        dim3 grid((N + BN - 1) / BN, (M + BM - 1) / BM);
        if (bt) {
            gemm_wmma_kernel<0, 1><<<grid, GTHREADS, 0, stream>>>(A, Bm, bias, C, M, N, Kd, Kd, ldbK, ldbN, N);
        } else if (epi == 1) {
            gemm_wmma_kernel<1, 0><<<grid, GTHREADS, 0, stream>>>(A, Bm, bias, C, M, N, Kd, Kd, ldbK, ldbN, N);
        } else if (epi == 2) {
            gemm_wmma_kernel<2, 0><<<grid, GTHREADS, 0, stream>>>(A, Bm, bias, C, M, N, Kd, Kd, ldbK, ldbN, N);
        } else {
            gemm_wmma_kernel<0, 0><<<grid, GTHREADS, 0, stream>>>(A, Bm, bias, C, M, N, Kd, Kd, ldbK, ldbN, N);
        }
    };

    auto mha = [&](const float* qin, int tq, const float* kin, int tk,
                   int m, float lam_init, int causal, float* outbuf) {
        gemm(qin, F(m + 7), nullptr, qb, BATCH * tq, HID, HID, HID, 1, 0, 0); // wq
        gemm(kin, F(m + 5), nullptr, kb, BATCH * tk, HID, HID, HID, 1, 0, 0); // wk
        gemm(kin, F(m + 8), nullptr, vb, BATCH * tk, HID, HID, HID, 1, 0, 0); // wv
        dim3 g((tq + 15) / 16, NHD, BATCH);
        diff_attn_kernel<<<g, 32, 0, stream>>>(qb, kb, vb,
            F(m + 2), F(m + 3), F(m + 0), F(m + 1), F(m + 4),
            ao, tq, tk, causal, lam_init);
        gemm(ao, F(m + 6), nullptr, outbuf, BATCH * tq, HID, HID, HID, 1, 0, 0); // wo
    };

    auto lam_init_fn = [](int depth) { return 0.8f - 0.6f * expf(-0.3f * (float)depth); };

    // ---- keyword branch: kw = diff_mha(We[kwtok]) ----
    {
        int rows = BATCH * KWN;
        embed_kw_kernel<<<((long)rows * HID + 255) / 256, 256, 0, stream>>>(kwtok, F(IDX_We), kw, rows);
        mha(kw, KWN, kw, KWN, LANG, lam_init_fn(0), 0, kw);
    }

    // ---- token embeddings + positions ----
    {
        int rows = BATCH * TSEQ;
        embed_tok_kernel<<<((long)rows * HID + 255) / 256, 256, 0, stream>>>(tokens, F(IDX_We), F(IDX_wpe), x, rows);
    }

    // ---- ef = vis @ vf_w + vf_b ----
    gemm(vis, F(IDX_vfw), F(IDX_vfb), ef, BATCH * NPV, HID, ENCD, HID, 1, 1, 0);

    // ---- fuser stack ----
    for (int i = 0; i < NLAY; ++i) {
        int base = FUS0 + i * 15;
        float li = lam_init_fn(i);
        if (i == 0) mha(ef, NPV, kw, KWN, base + 0, li, 0, t2);
        else        mha(ef, NPV, ef, NPV, base + 0, li, 0, t2);
        add_ln_kernel<<<BATCH * NPV, 128, 0, stream>>>(ef, t2, F(base + 10), F(base + 9));   // ln1
        gemm(ef, F(base + 13), nullptr, t1, BATCH * NPV, FCD, HID, FCD, 1, 2, 0);            // fc + gelu
        gemm(t1, F(base + 14), nullptr, t2, BATCH * NPV, HID, FCD, HID, 1, 0, 0);            // proj
        add_ln_kernel<<<BATCH * NPV, 128, 0, stream>>>(ef, t2, F(base + 12), F(base + 11));  // ln2
    }

    // ---- decoder stack ----
    for (int i = 0; i < NLAY; ++i) {
        int base = DEC0 + i * 26;
        float li = lam_init_fn(i);
        mha(x, TSEQ, x, TSEQ, base + 17, li, 1, t2);                                          // sa (causal)
        add_ln_kernel<<<BATCH * TSEQ, 128, 0, stream>>>(x, t2, F(base + 10), F(base + 9));    // ln1
        mha(x, TSEQ, ef, NPV, base + 0, li, 0, t2);                                           // ca
        add_ln_kernel<<<BATCH * TSEQ, 128, 0, stream>>>(x, t2, F(base + 12), F(base + 11));   // ln2
        gemm(x, F(base + 15), nullptr, t1, BATCH * TSEQ, FCD, HID, FCD, 1, 2, 0);             // fc + gelu
        gemm(t1, F(base + 16), nullptr, t2, BATCH * TSEQ, HID, FCD, HID, 1, 0, 0);            // proj
        add_ln_kernel<<<BATCH * TSEQ, 128, 0, stream>>>(x, t2, F(base + 14), F(base + 13));   // ln3
    }

    // ---- logits = x @ We^T  (B stored K-contiguous: B(k,n) = We[n*HID + k]) ----
    gemm(x, F(IDX_We), nullptr, (float*)d_out, BATCH * TSEQ, VOC, HID, 1, HID, 0, 1);
}
